// ExpertChoiceMoEMatcher_58248346468718
// MI455X (gfx1250) — compile-verified, matrix operand order checked by reading, not run on hardware
//
#include <hip/hip_runtime.h>
#include <hip/hip_bf16.h>
#include <cmath>

// ---------------------------------------------------------------------------
// MI455X (gfx1250) ExpertChoice MoE:
//   scores = x(65536x512) @ G(512x64)            -> V_WMMA_F32_16X16X4_F32
//   per-expert top-k (K=2048) radix select
//   per-expert complex GEMM (2048x256 @ 256x256) -> V_WMMA_F32_16X16X32_BF16
//        fp32 emulated via bf16 hi/lo split, 3-term products, f32 accum
//   token gather staged via GLOBAL_LOAD_ASYNC_TO_LDS_B128 (ASYNCcnt path)
//   scatter-add (global_atomic_add_f32), count-normalize, exact GELU
// ---------------------------------------------------------------------------

typedef float v2f  __attribute__((ext_vector_type(2)));
typedef float v8f  __attribute__((ext_vector_type(8)));
typedef __bf16 v16bf __attribute__((ext_vector_type(16)));
typedef int   v4ig __attribute__((vector_size(4 * sizeof(int))));  // builtin's b128 payload type

union BFx16 { v16bf v; unsigned short s[16]; uint4 q[2]; };

#define WMMA_BF16(A, B, C) \
  __builtin_amdgcn_wmma_f32_16x16x32_bf16(false, (A), false, (B), (short)0, (C), false, false)
#define WMMA_F32(A, B, C) \
  __builtin_amdgcn_wmma_f32_16x16x4_f32(false, (A), false, (B), (short)0, (C), false, false)

#if defined(__has_builtin)
#if __has_builtin(__builtin_amdgcn_global_load_async_to_lds_b128)
#define HAVE_ASYNC_LDS 1
#endif
#endif

// ---- helpers --------------------------------------------------------------
__device__ __forceinline__ unsigned short f32_to_bf16_rne(float f) {
  unsigned u = __float_as_uint(f);
  unsigned r = u + 0x7FFFu + ((u >> 16) & 1u);
  return (unsigned short)(r >> 16);
}
__device__ __forceinline__ float bf16_to_f32(unsigned short h) {
  return __uint_as_float(((unsigned)h) << 16);
}
__device__ __forceinline__ void split_bf16(float v, unsigned short& hi, unsigned short& lo) {
  hi = f32_to_bf16_rne(v);
  lo = f32_to_bf16_rne(v - bf16_to_f32(hi));
}
// order-preserving unsigned key for float compare (matches top_k ordering)
__device__ __forceinline__ unsigned mono_key(float f) {
  unsigned u = __float_as_uint(f);
  return (u & 0x80000000u) ? ~u : (u | 0x80000000u);
}

static constexpr int BT = 65536;   // tokens
static constexpr int D  = 256;     // features (complex)
static constexpr int E  = 64;      // experts
static constexpr int KN = 2048;    // top-k per expert

// ---------------------------------------------------------------------------
// Kernel P: pack expert weights into bf16 hi/lo, WMMA-B operand lane order.
// layout: [e][c(r/i)][kb(8)][ftile(16)][lane(32)][16 bf16] ; per-lane 32B blob
// B-tile (32x16 bf16): lane L -> N = L%16 ; elem j -> K = (j<8?j:j+8)+8*(L/16)
// ---------------------------------------------------------------------------
__global__ __launch_bounds__(256) void packw_kernel(
    const float* __restrict__ W, unsigned short* __restrict__ whi,
    unsigned short* __restrict__ wlo) {
  unsigned gid = blockIdx.x * 256u + threadIdx.x;   // 64*2*8*16*32 = 524288
  unsigned lane = gid & 31u;
  unsigned ft   = (gid >> 5) & 15u;
  unsigned kb   = (gid >> 9) & 7u;
  unsigned c    = (gid >> 12) & 1u;
  unsigned e    = gid >> 13;
  unsigned half = lane >> 4, lm = lane & 15u;
  unsigned f    = ft * 16u + lm;
  BFx16 hb, lb;
#pragma unroll
  for (int j = 0; j < 16; ++j) {
    unsigned kk = kb * 32u + (unsigned)((j < 8) ? j : j + 8) + 8u * half;
    float v = W[(((size_t)e * D + kk) * D + f) * 2 + c];
    split_bf16(v, hb.s[j], lb.s[j]);
  }
  size_t base = ((((size_t)((e * 2u + c) * 8u + kb)) * 16u + ft) * 32u + lane) * 16u;
  ((uint4*)(whi + base))[0] = hb.q[0];
  ((uint4*)(whi + base))[1] = hb.q[1];
  ((uint4*)(wlo + base))[0] = lb.q[0];
  ((uint4*)(wlo + base))[1] = lb.q[1];
}

// ---------------------------------------------------------------------------
// Kernel 1: gate GEMM, scoresT[e][t] = sum_k x[t][k] * G[k][e]  (K=512)
// 512 blocks x 256 thr; wave w handles 16 tokens, 4 N-tiles (64 experts).
// A 16x4 f32: lane L -> M=L%16 ; elem j -> K = j + 2*(L/16).  B symmetric.
// ---------------------------------------------------------------------------
__global__ __launch_bounds__(256) void gate_kernel(
    const float* __restrict__ x, const float* __restrict__ gw,
    float* __restrict__ scoresT) {
  __shared__ float sT[64][129];        // 33KB: [expert][token-local]
  const int tid = threadIdx.x, lane = tid & 31, wave = tid >> 5;
  const int half = lane >> 4, lm = lane & 15;
  const int tloc0 = wave * 16;
  const size_t trow = (size_t)(blockIdx.x * 128 + tloc0 + lm);
  const float2* xr2 = (const float2*)(x + trow * 512);
  v8f acc[4] = {};
  for (int k = 0; k < 512; k += 4) {
    float2 av = xr2[(k >> 1) + half];
    v2f a; a.x = av.x; a.y = av.y;
#pragma unroll
    for (int nt = 0; nt < 4; ++nt) {
      v2f b;
      b.x = gw[(k + 2 * half) * 64 + nt * 16 + lm];
      b.y = gw[(k + 2 * half + 1) * 64 + nt * 16 + lm];
      acc[nt] = WMMA_F32(a, b, acc[nt]);
    }
  }
  // C layout: VGPR r -> M = r + 8*half ; N = lm. Transpose through LDS.
#pragma unroll
  for (int nt = 0; nt < 4; ++nt)
#pragma unroll
    for (int r = 0; r < 8; ++r)
      sT[nt * 16 + lm][tloc0 + r + 8 * half] = acc[nt][r];
  __syncthreads();
  for (int i = tid; i < 64 * 128; i += 256) {
    int e = i >> 7, tl = i & 127;
    scoresT[(size_t)e * BT + blockIdx.x * 128 + tl] = sT[e][tl];
  }
}

// ---------------------------------------------------------------------------
// Kernel 2: per-expert top-K radix select (4x8-bit passes) + compaction.
// One 1024-thread block per expert.
// ---------------------------------------------------------------------------
__global__ __launch_bounds__(1024) void topk_kernel(
    const float* __restrict__ scoresT, int* __restrict__ oidx,
    float* __restrict__ oscore, float* __restrict__ counts) {
  const int e = blockIdx.x;
  const float* sc = scoresT + (size_t)e * BT;
  __shared__ unsigned hist[256];
  __shared__ unsigned s_prefix, s_remaining, s_cnt_gt, s_cnt_eq;
  const int tid = threadIdx.x;
  if (tid == 0) { s_prefix = 0; s_remaining = KN; s_cnt_gt = 0; s_cnt_eq = 0; }
  for (int pass = 0; pass < 4; ++pass) {
    const int shift = 24 - pass * 8;
    if (tid < 256) hist[tid] = 0;
    __syncthreads();
    const unsigned prefix = s_prefix;
    const unsigned himask = (pass == 0) ? 0u : (0xFFFFFFFFu << (shift + 8));
    for (int i = tid; i < BT; i += 1024) {
      unsigned k = mono_key(sc[i]);
      if ((k & himask) == prefix) atomicAdd(&hist[(k >> shift) & 255u], 1u);
    }
    __syncthreads();
    if (tid == 0) {
      unsigned rem = s_remaining, bin = 0;
      for (int b = 255; b >= 0; --b) {
        unsigned c = hist[b];
        if (c >= rem) { bin = (unsigned)b; break; }
        rem -= c;
      }
      s_prefix = prefix | (bin << shift);
      s_remaining = rem;
    }
    __syncthreads();
  }
  const unsigned T = s_prefix;           // threshold key
  const unsigned need_eq = s_remaining;  // how many ==T to take
  for (int i = tid; i < BT; i += 1024) {
    float v = sc[i];
    unsigned k = mono_key(v);
    if (k > T) {
      unsigned p = atomicAdd(&s_cnt_gt, 1u);
      oidx[e * KN + p] = i; oscore[e * KN + p] = v;
      unsafeAtomicAdd(&counts[i], 1.0f);
    } else if (k == T) {
      unsigned q = atomicAdd(&s_cnt_eq, 1u);
      if (q < need_eq) {
        unsigned p = (unsigned)KN - need_eq + q;
        oidx[e * KN + p] = i; oscore[e * KN + p] = v;
        unsafeAtomicAdd(&counts[i], 1.0f);
      }
    }
  }
}

// ---------------------------------------------------------------------------
// Kernel 3: expert complex GEMM, bf16x3 split, f32 accum, scaled scatter-add.
// grid = 64 experts x 64 tiles of 32 tokens ; 256 thr (8 waves).
// wave: mt = w&1 (16-token M-tile), nq = w>>1 (64-col f quarter, 4 f-tiles).
// A 16x32 bf16: lane L -> M=L%16 ; elem j -> K=(j<8?j:j+8)+8*(L/16).
// Gather uses async global->LDS (ASYNCcnt) when the builtin is available.
// ---------------------------------------------------------------------------
__global__ __launch_bounds__(256) void expert_kernel(
    const float* __restrict__ x, const unsigned short* __restrict__ whi,
    const unsigned short* __restrict__ wlo, const int* __restrict__ tidx,
    const float* __restrict__ tscore, float* __restrict__ outacc) {
  const int e  = blockIdx.x >> 6;
  const int kt = blockIdx.x & 63;
  __shared__ float2 xg[32 * 256];      // 64KB gathered tokens (r,i interleaved)
  const int tid = threadIdx.x;
  for (int i = tid; i < 32 * 128; i += 256) {   // float4 granules
    int row = i >> 7, off = i & 127;
    int t = tidx[e * KN + kt * 32 + row];
    const float4* src = (const float4*)(x + (size_t)t * 512) + off;
    float4* dst = (float4*)(xg + row * 256) + off;
#if HAVE_ASYNC_LDS
    __builtin_amdgcn_global_load_async_to_lds_b128(
        (v4ig*)(void*)src, (v4ig*)(void*)dst, 0, 0);
#else
    *dst = *src;
#endif
  }
#if HAVE_ASYNC_LDS
#if __has_builtin(__builtin_amdgcn_s_wait_asynccnt)
  __builtin_amdgcn_s_wait_asynccnt(0);
#else
  asm volatile("s_wait_asynccnt 0x0" ::: "memory");
#endif
#endif
  __syncthreads();
  const int lane = tid & 31, wave = tid >> 5;
  const int half = lane >> 4, lm = lane & 15;
  const int mt = wave & 1, nq = wave >> 1;
  const int arow = mt * 16 + lm;
  v8f accR[4] = {}, accI[4] = {};
  for (int kb = 0; kb < 8; ++kb) {
    BFx16 arh, arl, aih, ail, anh, anl;   // real hi/lo, imag hi/lo, -imag hi/lo
#pragma unroll
    for (int j = 0; j < 16; ++j) {
      int kk = kb * 32 + ((j < 8) ? j : j + 8) + 8 * half;
      float2 v = xg[arow * 256 + kk];
      split_bf16(v.x, arh.s[j], arl.s[j]);
      split_bf16(v.y, aih.s[j], ail.s[j]);
      anh.s[j] = aih.s[j] ^ 0x8000u;      // bf16 negate (f16/bf16 NEG is C-only)
      anl.s[j] = ail.s[j] ^ 0x8000u;
    }
    if (kb + 1 < 8)   // warm L2->L0 for next k-block's B operands
      __builtin_prefetch(whi + ((((size_t)((e * 2) * 8 + kb + 1)) * 16 + nq * 4) * 32 + lane) * 16, 0, 1);
#pragma unroll
    for (int ft = 0; ft < 4; ++ft) {
      const int ftg = nq * 4 + ft;
      size_t br = ((((size_t)((e * 2 + 0) * 8 + kb)) * 16 + ftg) * 32 + lane) * 16;
      size_t bi = ((((size_t)((e * 2 + 1) * 8 + kb)) * 16 + ftg) * 32 + lane) * 16;
      BFx16 brh, brl, bih, bil;
      brh.q[0] = ((const uint4*)(whi + br))[0]; brh.q[1] = ((const uint4*)(whi + br))[1];
      brl.q[0] = ((const uint4*)(wlo + br))[0]; brl.q[1] = ((const uint4*)(wlo + br))[1];
      bih.q[0] = ((const uint4*)(whi + bi))[0]; bih.q[1] = ((const uint4*)(whi + bi))[1];
      bil.q[0] = ((const uint4*)(wlo + bi))[0]; bil.q[1] = ((const uint4*)(wlo + bi))[1];
      // yr += xr*wr - xi*wi   (3-term bf16x3 per product)
      accR[ft] = WMMA_BF16(arh.v, brh.v, accR[ft]);
      accR[ft] = WMMA_BF16(arh.v, brl.v, accR[ft]);
      accR[ft] = WMMA_BF16(arl.v, brh.v, accR[ft]);
      accR[ft] = WMMA_BF16(anh.v, bih.v, accR[ft]);
      accR[ft] = WMMA_BF16(anh.v, bil.v, accR[ft]);
      accR[ft] = WMMA_BF16(anl.v, bih.v, accR[ft]);
      // yi += xr*wi + xi*wr
      accI[ft] = WMMA_BF16(arh.v, bih.v, accI[ft]);
      accI[ft] = WMMA_BF16(arh.v, bil.v, accI[ft]);
      accI[ft] = WMMA_BF16(arl.v, bih.v, accI[ft]);
      accI[ft] = WMMA_BF16(aih.v, brh.v, accI[ft]);
      accI[ft] = WMMA_BF16(aih.v, brl.v, accI[ft]);
      accI[ft] = WMMA_BF16(ail.v, brh.v, accI[ft]);
    }
  }
  // C layout: VGPR r -> M = r + 8*half ; N = lm. Scale by gate score, scatter.
#pragma unroll
  for (int r = 0; r < 8; ++r) {
    const int row = mt * 16 + r + 8 * half;
    const int gk = e * KN + kt * 32 + row;
    const int t = tidx[gk];
    const float wgt = tscore[gk];
#pragma unroll
    for (int ft = 0; ft < 4; ++ft) {
      const int f = nq * 64 + ft * 16 + lm;
      const size_t o = ((size_t)t * D + f) * 2;
      unsafeAtomicAdd(&outacc[o],     accR[ft][r] * wgt);
      unsafeAtomicAdd(&outacc[o + 1], accI[ft][r] * wgt);
    }
  }
}

// ---------------------------------------------------------------------------
// Kernel 4: h = acc / max(count,1); res = gelu_exact(h + bias)
// ---------------------------------------------------------------------------
__global__ __launch_bounds__(256) void finalize_kernel(
    float* __restrict__ res, const float* __restrict__ counts,
    const float* __restrict__ bias) {
  size_t gid = (size_t)blockIdx.x * 256 + threadIdx.x;   // one (t,d) pair each
  int t = (int)(gid >> 8), d = (int)(gid & 255);
  float c = fmaxf(counts[t], 1.0f);
  float inv = 1.0f / c;
  float bb = bias[d];
  float2 h = ((float2*)res)[gid];
  float a0 = h.x * inv + bb, a1 = h.y * inv + bb;
  const float is2 = 0.70710678118654752440f;
  h.x = 0.5f * a0 * (1.0f + erff(a0 * is2));
  h.y = 0.5f * a1 * (1.0f + erff(a1 * is2));
  ((float2*)res)[gid] = h;
}

// ---------------------------------------------------------------------------
extern "C" void kernel_launch(void* const* d_in, const int* in_sizes, int n_in,
                              void* d_out, int out_size, void* d_ws, size_t ws_size,
                              hipStream_t stream) {
  (void)in_sizes; (void)n_in; (void)out_size; (void)ws_size;
  const float* x    = (const float*)d_in[0];
  const float* gw   = (const float*)d_in[1];
  const float* W    = (const float*)d_in[2];
  const float* bias = (const float*)d_in[3];

  float* out    = (float*)d_out;
  float* res    = out;                                   // 33,554,432 f32
  int*   oidx   = (int*)(out + (size_t)BT * D * 2);      // 131,072 i32
  float* oscore = out + (size_t)BT * D * 2 + E * KN;     // 131,072 f32
  float* counts = oscore + E * KN;                       // 65,536  f32

  // workspace: scoresT (16MB) | whi (16MB) | wlo (16MB)  => 48MB total
  float* scoresT = (float*)d_ws;
  unsigned short* whi = (unsigned short*)((char*)d_ws + (size_t)E * BT * 4);
  unsigned short* wlo = whi + (size_t)E * 2 * D * D;

  (void)hipMemsetAsync(res, 0, (size_t)BT * D * 2 * sizeof(float), stream);
  (void)hipMemsetAsync(counts, 0, (size_t)BT * sizeof(float), stream);

  packw_kernel<<<2048, 256, 0, stream>>>(W, whi, wlo);
  gate_kernel<<<512, 256, 0, stream>>>(x, gw, scoresT);
  topk_kernel<<<E, 1024, 0, stream>>>(scoresT, oidx, oscore, counts);
  expert_kernel<<<E * 64, 256, 0, stream>>>(x, whi, wlo, oidx, oscore, res);
  finalize_kernel<<<BT * D / 256, 256, 0, stream>>>(res, counts, bias);
}
